// CustomLoss_50843822850472
// MI455X (gfx1250) — compile-verified, matrix operand order checked by reading, not run on hardware
//
#include <hip/hip_runtime.h>

#define D 128
// exp(s/tau) = exp2(s * log2(e)/tau); log2(e)/0.07 = 20.609929155556618
#define EXP2_COEF 20.60992915555662f
#define EPS 1e-10f

// Pair-block geometry: 256x256 sim blocks, staged as two 128-column phases.
// LDS chunk rows padded to 136 halves (272 B = 68 dwords; 68 % 64 = 4 -> the
// 16 row-striding lanes of a ds_load_b128 hit 16 distinct 4-bank groups).
#define RBLK   256
#define PHROWS 128
#define STRIDE_H 136

typedef __attribute__((ext_vector_type(16))) _Float16 v16h;
typedef __attribute__((ext_vector_type(8)))  _Float16 v8h;
typedef __attribute__((ext_vector_type(8)))  float    v8f;

// Raw v_exp_f32: our arguments are bounded (|s|*20.61 <= ~21), far from the
// denormal range that OCML's exp2f guards against with cmp/add/ldexp.
#if __has_builtin(__builtin_amdgcn_exp2f)
#define fast_exp2(x) __builtin_amdgcn_exp2f(x)
#else
#define fast_exp2(x) exp2f(x)
#endif

#if __has_builtin(__builtin_amdgcn_global_load_async_to_lds_b128)
#define HAVE_ASYNC_LDS 1
// The builtin expects pointers to a 16-byte int vector in addrspace(1)/(3).
typedef int vi4 __attribute__((vector_size(16)));
typedef __attribute__((address_space(1))) vi4 g_vi4;
typedef __attribute__((address_space(3))) vi4 l_vi4;
// Flat LDS addresses keep the byte offset in the low 32 bits, so integer
// truncation yields a valid as(3) pointer.
__device__ __forceinline__ l_vi4* as_lds(void* p) {
  return (l_vi4*)(unsigned)(unsigned long long)p;
}
__device__ __forceinline__ g_vi4* as_gbl(const void* p) {
  return (g_vi4*)(unsigned long long)p;
}
#else
#define HAVE_ASYNC_LDS 0
#endif

// ---------------------------------------------------------------------------
// K1: L2-normalize rows of fp32 features -> f16 rows in workspace.
// ---------------------------------------------------------------------------
__global__ __launch_bounds__(256) void normalize_rows(const float* __restrict__ feat,
                                                      _Float16* __restrict__ fn,
                                                      int N) {
  const int row  = blockIdx.x * 8 + (threadIdx.x >> 5);
  const int lane = threadIdx.x & 31;
  if (row >= N) return;

  const float4 v = ((const float4*)(feat + (size_t)row * D))[lane];
  float ss = v.x * v.x + v.y * v.y + v.z * v.z + v.w * v.w;
#pragma unroll
  for (int m = 16; m >= 1; m >>= 1) ss += __shfl_xor(ss, m, 32);

  const float inv = 1.0f / fmaxf(sqrtf(ss), 1e-12f);
  _Float16* q = fn + (size_t)row * D + lane * 4;
  q[0] = (_Float16)(v.x * inv);
  q[1] = (_Float16)(v.y * inv);
  q[2] = (_Float16)(v.z * inv);
  q[3] = (_Float16)(v.w * inv);
}

// ---------------------------------------------------------------------------
// Fragment gather for v_wmma_f32_16x16x32_f16 (wave32), 16-bit 16x32 layout:
// lane<16 -> M=lane, halves 0..7 = K 0..7, 8..15 = K 16..23;
// lane>=16 -> M=lane-16, halves 0..7 = K 8..15, 8..15 = K 24..31.
// ---------------------------------------------------------------------------
__device__ __forceinline__ v16h gather_frag(const v8h* __restrict__ p, int kk, int half) {
  v8h lo = p[kk * 4 + half];
  v8h hi = p[kk * 4 + half + 2];
  v16h a;
#pragma unroll
  for (int t = 0; t < 8; ++t) { a[t] = lo[t]; a[t + 8] = hi[t]; }
  return a;
}

// ---------------------------------------------------------------------------
// K2: symmetric pair-block kernel. Block handles sim block (bi,bj), bi<=bj,
// of 256x256. Row sums accumulate as usual; for off-diagonal pairs the
// transposed (column) sums are also credited to rows of bj, halving total
// WMMA + exp work. Columns staged via async global->LDS in two 128-row
// phases; waves rotate over column tiles so the LDS column accumulators are
// updated race-free and deterministically (barrier per round, no atomics).
// ---------------------------------------------------------------------------
__global__ __launch_bounds__(512) void simloss_pairs(const _Float16* __restrict__ fn,
                                                     const float* __restrict__ alpha_p,
                                                     float* __restrict__ pos_part,
                                                     float* __restrict__ all_part,
                                                     int N) {
  __shared__ _Float16 sB[PHROWS * STRIDE_H];   // 34816 B
  __shared__ float col_acc[2][PHROWS][2];      // 2048 B: [wave-group][col][pos/all]

  const int NB = N / RBLK;  // 32
  // Decode pair (bi, bj) from linear block id: C(b) = b*NB - b*(b-1)/2.
  int bi = 0;
  {
    const int p = blockIdx.x;
    while (((bi + 1) * NB - ((bi + 1) * bi) / 2) <= p) ++bi;
  }
  const int bj = bi + (blockIdx.x - (bi * NB - (bi * (bi - 1)) / 2));
  const bool offdiag = (bi != bj);   // block-uniform

  const int tid  = threadIdx.x;
  const int lane = tid & 31;
  const int half = lane >> 4;
  const int rr   = lane & 15;
  const int wave = tid >> 5;          // 0..15
  const int wgrp = wave >> 3;         // 0/1: two groups sweep the same tiles
  const int wrot = wave & 7;
  const int row_base = bi * RBLK + wave * 16;
  const float alpha = *alpha_p;

  // Loop-invariant A fragments: 16 rows x K=128 (4 k-steps), from global.
  v16h a[4];
  {
    const v8h* ap = (const v8h*)(fn + (size_t)(row_base + rr) * D);
#pragma unroll
    for (int kk = 0; kk < 4; ++kk) a[kk] = gather_frag(ap, kk, half);
  }

  float acc_pos[8], acc_all[8];
#pragma unroll
  for (int r = 0; r < 8; ++r) { acc_pos[r] = 0.0f; acc_all[r] = 0.0f; }

  // Staging map: 128 rows x 16 b128-segments = 2048 transfers / 512 threads.
  const int srow0 = tid >> 4;  // 0..31, +32 per step (4 steps)
  const int sseg  = tid & 15;

  for (int phase = 0; phase < 2; ++phase) {
    const int col0 = bj * RBLK + phase * PHROWS;

    // Zero column accumulators (512 floats, one per thread).
    ((float*)col_acc)[tid] = 0.0f;

    // Stage this phase's 128 B-columns into LDS.
#pragma unroll
    for (int q = 0; q < 4; ++q) {
      const int row = srow0 + q * 32;
      const _Float16* g = fn + (size_t)(col0 + row) * D + sseg * 8;
      _Float16* l = &sB[row * STRIDE_H + sseg * 8];
#if HAVE_ASYNC_LDS
      __builtin_amdgcn_global_load_async_to_lds_b128(as_gbl(g), as_lds(l), 0, 0);
#else
      *(uint4*)l = *(const uint4*)g;
#endif
    }
#if HAVE_ASYNC_LDS
    asm volatile("s_wait_asynccnt 0x0" ::: "memory");
#endif
    __syncthreads();

    // 8 rounds; within each round, the 8 waves of a group cover the 8 column
    // tiles without overlap -> race-free col_acc updates.
    for (int s = 0; s < 8; ++s) {
      const int tt = (wrot + s) & 7;
      const int col_base = col0 + tt * 16;
      const v8h* lp = (const v8h*)(&sB[(tt * 16 + rr) * STRIDE_H]);
      v16h b[4];
#pragma unroll
      for (int kk = 0; kk < 4; ++kk) b[kk] = gather_frag(lp, kk, half);

      v8f acc = {};
#pragma unroll
      for (int kk = 0; kk < 4; ++kk)
        acc = __builtin_amdgcn_wmma_f32_16x16x32_f16(false, a[kk], false, b[kk],
                                                     (short)0, acc, false, false);

      const int gcol = col_base + rr;
      if (offdiag) {
        // No diagonal possible here: skip the mask, keep column sums.
        float cs_pos = 0.0f, cs_all = 0.0f;
#pragma unroll
        for (int r = 0; r < 8; ++r) {
          const float sv = acc[r];
          const float e  = fast_exp2(sv * EXP2_COEF);    // exp(sv/tau)
          const float ep = (sv >= alpha) ? e : 0.0f;     // sim >= alpha branch
          acc_all[r] += e;
          acc_pos[r] += ep;
          cs_all += e;
          cs_pos += ep;
        }
        // Merge the two halves of this column, then one lane per column
        // accumulates into LDS (unique column per wave-group this round).
        cs_all += __shfl_xor(cs_all, 16, 32);
        cs_pos += __shfl_xor(cs_pos, 16, 32);
        if (half == 0) {
          const int c = tt * 16 + rr;
          col_acc[wgrp][c][0] += cs_pos;
          col_acc[wgrp][c][1] += cs_all;
        }
      } else {
        // Diagonal block: mask out sim(i,i); no column sums (row sums cover it).
#pragma unroll
        for (int r = 0; r < 8; ++r) {
          const float sv = acc[r];
          float e = fast_exp2(sv * EXP2_COEF);
          const int grow = row_base + r + (half << 3);
          e = (gcol == grow) ? 0.0f : e;
          acc_all[r] += e;
          acc_pos[r] += (sv >= alpha) ? e : 0.0f;
        }
      }
      __syncthreads();
    }

    // Column (transposed) sums -> rows of bj, slot bi.
    if (offdiag && tid < PHROWS) {
      const int row = col0 + tid;
      pos_part[(size_t)bi * N + row] = col_acc[0][tid][0] + col_acc[1][tid][0];
      all_part[(size_t)bi * N + row] = col_acc[0][tid][1] + col_acc[1][tid][1];
    }
    __syncthreads();  // col_acc + sB reuse safe for next phase
  }

  // Row sums: reduce across the 16 lanes sharing each matrix row.
#pragma unroll
  for (int r = 0; r < 8; ++r) {
#pragma unroll
    for (int m = 8; m >= 1; m >>= 1) {
      acc_pos[r] += __shfl_xor(acc_pos[r], m, 32);
      acc_all[r] += __shfl_xor(acc_all[r], m, 32);
    }
  }
  if (rr == 0) {
#pragma unroll
    for (int r = 0; r < 8; ++r) {
      const int row = row_base + r + (half << 3);
      pos_part[(size_t)bj * N + row] = acc_pos[r];
      all_part[(size_t)bj * N + row] = acc_all[r];
    }
  }
}

// ---------------------------------------------------------------------------
// K3: deterministic single-block reduction -> mean loss.
// Every slot s in [0, NB) is written exactly once per row:
//   s>b: row sums of pair (b,s); s==b: diagonal pair; s<b: col sums of (s,b).
// ---------------------------------------------------------------------------
__global__ __launch_bounds__(1024) void finalize_loss(const float* __restrict__ pos_part,
                                                      const float* __restrict__ all_part,
                                                      float* __restrict__ out,
                                                      int N) {
  const int NB = N / RBLK;
  float local = 0.0f;
  for (int row = threadIdx.x; row < N; row += 1024) {
    float p = 0.0f, al = 0.0f;
    for (int s = 0; s < NB; ++s) {
      p  += pos_part[(size_t)s * N + row];
      al += all_part[(size_t)s * N + row];
    }
    local += __logf(al + 2.0f * EPS) - __logf(p + EPS);
  }

#pragma unroll
  for (int m = 16; m >= 1; m >>= 1) local += __shfl_xor(local, m, 32);

  __shared__ float smem[32];
  const int lane = threadIdx.x & 31, w = threadIdx.x >> 5;
  if (lane == 0) smem[w] = local;
  __syncthreads();
  if (w == 0) {
    float v = smem[lane];
#pragma unroll
    for (int m = 16; m >= 1; m >>= 1) v += __shfl_xor(v, m, 32);
    if (lane == 0) out[0] = v / (float)N;
  }
}

// ---------------------------------------------------------------------------
extern "C" void kernel_launch(void* const* d_in, const int* in_sizes, int n_in,
                              void* d_out, int out_size, void* d_ws, size_t ws_size,
                              hipStream_t stream) {
  const float* feat    = (const float*)d_in[0];
  const float* alpha_p = (const float*)d_in[1];
  const int N = in_sizes[0] / D;  // 8192

  _Float16* fn = (_Float16*)d_ws;                       // 2 MB f16 normalized
  const size_t fn_bytes = (size_t)N * D * sizeof(_Float16);
  float* pos_part = (float*)((char*)d_ws + fn_bytes);   // NB * N floats (1 MB)
  const int NB = N / RBLK;
  float* all_part = pos_part + (size_t)NB * N;          // NB * N floats (1 MB)
  float* out = (float*)d_out;

  normalize_rows<<<N / 8, 256, 0, stream>>>(feat, fn, N);
  const int npairs = NB * (NB + 1) / 2;                 // 528
  simloss_pairs<<<npairs, 512, 0, stream>>>(fn, alpha_p, pos_part, all_part, N);
  finalize_loss<<<1, 1024, 0, stream>>>(pos_part, all_part, out, N);
}
